// HyperAttention_1898375544988
// MI455X (gfx1250) — matrix operand order repaired and checked by reading, not compiled, not run on hardware
//
#include <hip/hip_runtime.h>
#include <math.h>

// ---- problem constants (B=2, H=16, N=8192, D=64) ----
#define BH     32          // B*H
#define NSEQ   8192
#define DIM    64
#define NPROJ  7
#define BLK    256         // LSH key block size
#define SAMP   256         // sampled residual keys
#define QTILE  128         // queries per attention workgroup
#define CHUNK  32          // keys per inner chunk
#define LOG32  3.4657359027997265f

typedef __bf16 bf16;
typedef __attribute__((ext_vector_type(16))) __bf16 v16bf;
typedef __attribute__((ext_vector_type(8)))  __bf16 v8bf;
typedef __attribute__((ext_vector_type(8)))  float  v8f;

__device__ __forceinline__ bf16 f2bf(float x) { return (bf16)x; }

// ---------------------------------------------------------------------------
// All-VALU 16-lane butterfly reductions via DPP16 (no ds_bpermute traffic).
// Steps: quad xor1 (0xB1), quad xor2 (0x4E), row_half_mirror (0x141),
// row_mirror (0x140).  After each step all lanes of the pairing group hold
// the partial, so the mirror steps complete the 16-lane reduction.
// ---------------------------------------------------------------------------
__device__ __forceinline__ float dpp_max16(float x) {
  float t;
  t = __int_as_float(__builtin_amdgcn_update_dpp(0, __float_as_int(x), 0x0B1, 0xF, 0xF, true));
  x = fmaxf(x, t);
  t = __int_as_float(__builtin_amdgcn_update_dpp(0, __float_as_int(x), 0x04E, 0xF, 0xF, true));
  x = fmaxf(x, t);
  t = __int_as_float(__builtin_amdgcn_update_dpp(0, __float_as_int(x), 0x141, 0xF, 0xF, true));
  x = fmaxf(x, t);
  t = __int_as_float(__builtin_amdgcn_update_dpp(0, __float_as_int(x), 0x140, 0xF, 0xF, true));
  return fmaxf(x, t);
}
__device__ __forceinline__ float dpp_sum16(float x) {
  float t;
  t = __int_as_float(__builtin_amdgcn_update_dpp(0, __float_as_int(x), 0x0B1, 0xF, 0xF, true));
  x += t;
  t = __int_as_float(__builtin_amdgcn_update_dpp(0, __float_as_int(x), 0x04E, 0xF, 0xF, true));
  x += t;
  t = __int_as_float(__builtin_amdgcn_update_dpp(0, __float_as_int(x), 0x141, 0xF, 0xF, true));
  x += t;
  t = __int_as_float(__builtin_amdgcn_update_dpp(0, __float_as_int(x), 0x140, 0xF, 0xF, true));
  return x + t;
}

// ---------------------------------------------------------------------------
// Kernel 1: angular LSH hash.  Gray-code perm of the reference == i ^ (i>>1).
// ---------------------------------------------------------------------------
__global__ __launch_bounds__(256)
void lsh_hash_kernel(const float* __restrict__ x,
                     const float* __restrict__ proj,  // [DIM][NPROJ]
                     int* __restrict__ hash_out) {
  __shared__ float pr[DIM * NPROJ];
  const int tid = threadIdx.x;
  for (int i = tid; i < DIM * NPROJ; i += 256) pr[i] = proj[i];
  __syncthreads();
  const long long row = (long long)blockIdx.x * 256 + tid;  // < BH*NSEQ
  const float* xp = x + row * DIM;
  float acc[NPROJ];
#pragma unroll
  for (int r = 0; r < NPROJ; ++r) acc[r] = 0.f;
  for (int d = 0; d < DIM; ++d) {
    const float v = xp[d];
#pragma unroll
    for (int r = 0; r < NPROJ; ++r) acc[r] += v * pr[d * NPROJ + r];
  }
  int bits = 0;
#pragma unroll
  for (int r = 0; r < NPROJ; ++r) bits |= (acc[r] > 0.f) ? (1 << r) : 0;
  hash_out[row] = bits ^ (bits >> 1);   // binary-reflected Gray code
}

// ---------------------------------------------------------------------------
// Kernel 2: stable counting sort (128 buckets) per (b,h) row -> argsort index.
// blockIdx 0..31 sort q rows, 32..63 sort k rows.
// ---------------------------------------------------------------------------
__global__ __launch_bounds__(256)
void lsh_sort_kernel(const int* __restrict__ q_hash,
                     const int* __restrict__ k_hash,
                     int* __restrict__ q_idx,
                     int* __restrict__ k_idx) {
  const int which = blockIdx.x >> 5;
  const int bh    = blockIdx.x & 31;
  const int* h = (which ? k_hash : q_hash) + (long long)bh * NSEQ;
  int* idx     = (which ? k_idx  : q_idx)  + (long long)bh * NSEQ;

  __shared__ int cnt[128];
  __shared__ int base[128];
  __shared__ int ch[256];
  const int tid = threadIdx.x;

  if (tid < 128) cnt[tid] = 0;
  __syncthreads();
  for (int i = tid; i < NSEQ; i += 256) atomicAdd(&cnt[h[i]], 1);
  __syncthreads();
  if (tid == 0) {
    int run = 0;
    for (int b = 0; b < 128; ++b) { base[b] = run; run += cnt[b]; }
  }
  __syncthreads();
  if (tid < 128) cnt[tid] = 0;
  __syncthreads();
  // stable: process elements in index order, rank within bucket via scan
  for (int c = 0; c < NSEQ / 256; ++c) {
    const int i  = c * 256 + tid;
    const int hv = h[i];
    ch[tid] = hv;
    __syncthreads();
    int r = 0;
    for (int j = 0; j < tid; ++j) r += (ch[j] == hv) ? 1 : 0;
    idx[base[hv] + cnt[hv] + r] = i;
    __syncthreads();
    atomicAdd(&cnt[hv], 1);
    __syncthreads();
  }
}

// ---------------------------------------------------------------------------
// Kernel 3: fused block-diagonal + sampled-residual attention.
// One workgroup = 128 sorted queries of one (b,h); 8 waves x 16 query rows.
// Online softmax over 512 keys: chunks 0-7 diag block, 8-15 sampled (+log32,
// -inf on overlap).  Result scattered to original positions via q_idx.
// ---------------------------------------------------------------------------
__global__ __launch_bounds__(256)
void hyper_attn_kernel(const float* __restrict__ query,
                       const float* __restrict__ key,
                       const float* __restrict__ value,
                       const int*   __restrict__ sampled_set,
                       const int*   __restrict__ q_idx,
                       const int*   __restrict__ k_idx,
                       float* __restrict__ out) {
  __shared__ __align__(32) bf16 q_lds[QTILE][DIM];      // 16 KB
  __shared__ __align__(32) bf16 k_lds[CHUNK][DIM];      // 4 KB  [key][dim]
  __shared__ __align__(32) bf16 vT_lds[DIM][CHUNK];     // 4 KB  [dim][key]
  __shared__ __align__(32) bf16 p_lds[8][16][CHUNK];    // 8 KB  per-wave P
  __shared__ float bias_lds[CHUNK];

  const int tid  = threadIdx.x;
  const int lane = tid & 31;
  const int wave = tid >> 5;
  const int bh   = blockIdx.x >> 6;      // 64 query tiles per (b,h)
  const int tile = blockIdx.x & 63;
  const int blk  = tile >> 1;            // LSH block of these queries
  const long long bh_off = (long long)bh * NSEQ;
  const float scale = 0.125f;            // 64^-0.5

  // ---- stage Q tile (gather by q_idx, f32 -> bf16) ----
  {
    const int rrow = tid & 127;
    const int half = tid >> 7;
    const int qi   = q_idx[bh_off + tile * QTILE + rrow];
    const float* src = query + (bh_off + qi) * DIM + half * 32;
#pragma unroll
    for (int d = 0; d < 32; ++d) q_lds[rrow][half * 32 + d] = f2bf(src[d]);
  }
  __syncthreads();

  // per-wave Q A-operands (16x64, split in two K=32 slices)
  const int off  = (lane < 16) ? 0 : 16;      // A-layout half-wave K offset
  const int arow = wave * 16 + (lane & 15);
  v16bf qa0, qa1;
  {
    v8bf lo0 = *(const v8bf*)&q_lds[arow][0  + off];
    v8bf hi0 = *(const v8bf*)&q_lds[arow][8  + off];
    v8bf lo1 = *(const v8bf*)&q_lds[arow][32 + off];
    v8bf hi1 = *(const v8bf*)&q_lds[arow][40 + off];
    qa0 = __builtin_shufflevector(lo0, hi0, 0,1,2,3,4,5,6,7,8,9,10,11,12,13,14,15);
    qa1 = __builtin_shufflevector(lo1, hi1, 0,1,2,3,4,5,6,7,8,9,10,11,12,13,14,15);
  }

  v8f o0 = {}, o1 = {}, o2 = {}, o3 = {};
  float m[8], l[8];
#pragma unroll
  for (int r = 0; r < 8; ++r) { m[r] = -1e30f; l[r] = 0.f; }

#pragma unroll 2
  for (int c = 0; c < 16; ++c) {
    __syncthreads();
    // ---- stage 32-key K/V chunk (gathered, bf16; V transposed) ----
    {
      const int krow = tid >> 3;        // 0..31
      const int dseg = (tid & 7) * 8;   // 0..56
      int ki; float bias;
      if (c < 8) {                      // diagonal block keys
        ki = k_idx[bh_off + blk * BLK + c * CHUNK + krow];
        bias = 0.f;
      } else {                          // sampled residual keys
        const int ss = sampled_set[bh * SAMP + (c - 8) * CHUNK + krow];
        ki = k_idx[bh_off + ss];
        bias = ((ss >> 8) == blk) ? -1e30f : LOG32;   // overlap mask / Nk/S
      }
      if ((tid & 7) == 0) bias_lds[krow] = bias;
      const float* ksrc = key   + (bh_off + ki) * DIM + dseg;
      const float* vsrc = value + (bh_off + ki) * DIM + dseg;
#pragma unroll
      for (int d = 0; d < 8; ++d) {
        k_lds[krow][dseg + d]  = f2bf(ksrc[d]);
        vT_lds[dseg + d][krow] = f2bf(vsrc[d]);
      }
      // ---- L2 prefetch of next chunk's gathered rows (global_prefetch_b8)
      if (c + 1 < 16) {
        int ki2;
        if (c + 1 < 8) {
          ki2 = k_idx[bh_off + blk * BLK + (c + 1) * CHUNK + krow];
        } else {
          const int ss2 = sampled_set[bh * SAMP + (c + 1 - 8) * CHUNK + krow];
          ki2 = k_idx[bh_off + ss2];
        }
        __builtin_prefetch(key   + (bh_off + ki2) * DIM + dseg, 0, 3);
        __builtin_prefetch(value + (bh_off + ki2) * DIM + dseg, 0, 3);
      }
    }
    __syncthreads();

    // ---- S = Q K^T : two 16x16 tiles, K=64 via 2 WMMAs each ----
    v8f s0 = {}, s1 = {};
    {
      const int keyr = lane & 15;
      const int koff = (lane >> 4) << 4;   // B-layout half-wave K offset
      v16bf b00 = *(const v16bf*)&k_lds[keyr][koff];
      v16bf b01 = *(const v16bf*)&k_lds[keyr][32 + koff];
      v16bf b10 = *(const v16bf*)&k_lds[16 + keyr][koff];
      v16bf b11 = *(const v16bf*)&k_lds[16 + keyr][32 + koff];
      s0 = __builtin_amdgcn_wmma_f32_16x16x32_bf16(false, qa0, false, b00, (short)0, s0, false, false);
      s0 = __builtin_amdgcn_wmma_f32_16x16x32_bf16(false, qa1, false, b01, (short)0, s0, false, false);
      s1 = __builtin_amdgcn_wmma_f32_16x16x32_bf16(false, qa0, false, b10, (short)0, s1, false, false);
      s1 = __builtin_amdgcn_wmma_f32_16x16x32_bf16(false, qa1, false, b11, (short)0, s1, false, false);
    }

    // ---- online softmax (row stats via DPP16 butterflies, all-VALU) ----
    const float b0 = bias_lds[lane & 15];
    const float b1 = bias_lds[16 + (lane & 15)];
    float alpha[8];
#pragma unroll
    for (int r = 0; r < 8; ++r) {
      const float x0 = s0[r] * scale + b0;
      const float x1 = s1[r] * scale + b1;
      const float mx = dpp_max16(fmaxf(x0, x1));
      const float mn = fmaxf(m[r], mx);
      alpha[r] = __expf(m[r] - mn);
      const float p0 = __expf(x0 - mn);
      const float p1 = __expf(x1 - mn);
      const float rs = dpp_sum16(p0 + p1);
      l[r] = l[r] * alpha[r] + rs;
      m[r] = mn;
      s0[r] = p0; s1[r] = p1;
    }
#pragma unroll
    for (int r = 0; r < 8; ++r) {
      o0[r] *= alpha[r]; o1[r] *= alpha[r];
      o2[r] *= alpha[r]; o3[r] *= alpha[r];
    }

    // ---- C-layout P -> A-layout via per-wave LDS (same-wave, in-order DS) ----
    {
      const int prow = (lane < 16) ? 0 : 8;
      const int pcol = lane & 15;
#pragma unroll
      for (int r = 0; r < 8; ++r) {
        p_lds[wave][prow + r][pcol]      = f2bf(s0[r]);
        p_lds[wave][prow + r][16 + pcol] = f2bf(s1[r]);
      }
    }
    // ---- O += P (16x32) x V (32x64): 4 WMMAs over 16-col output tiles ----
    {
      v8bf plo = *(const v8bf*)&p_lds[wave][lane & 15][off];
      v8bf phi = *(const v8bf*)&p_lds[wave][lane & 15][off + 8];
      v16bf pa = __builtin_shufflevector(plo, phi, 0,1,2,3,4,5,6,7,8,9,10,11,12,13,14,15);
      const int dcol = lane & 15;
      const int kr   = (lane >> 4) << 4;
      v16bf vb0 = *(const v16bf*)&vT_lds[0  + dcol][kr];
      v16bf vb1 = *(const v16bf*)&vT_lds[16 + dcol][kr];
      v16bf vb2 = *(const v16bf*)&vT_lds[32 + dcol][kr];
      v16bf vb3 = *(const v16bf*)&vT_lds[48 + dcol][kr];
      o0 = __builtin_amdgcn_wmma_f32_16x16x32_bf16(false, pa, false, vb0, (short)0, o0, false, false);
      o1 = __builtin_amdgcn_wmma_f32_16x16x32_bf16(false, pa, false, vb1, (short)0, o1, false, false);
      o2 = __builtin_amdgcn_wmma_f32_16x16x32_bf16(false, pa, false, vb2, (short)0, o2, false, false);
      o3 = __builtin_amdgcn_wmma_f32_16x16x32_bf16(false, pa, false, vb3, (short)0, o3, false, false);
    }
  }

  // ---- normalize and scatter to original query positions (== unsort) ----
  {
    const int orow = wave * 16 + ((lane < 16) ? 0 : 8);
    const int ocol = lane & 15;
#pragma unroll
    for (int r = 0; r < 8; ++r) {
      const int qi = q_idx[bh_off + tile * QTILE + orow + r];
      const float inv = 1.f / l[r];
      float* dst = out + (bh_off + qi) * DIM;
      dst[0  + ocol] = o0[r] * inv;
      dst[16 + ocol] = o1[r] * inv;
      dst[32 + ocol] = o2[r] * inv;
      dst[48 + ocol] = o3[r] * inv;
    }
  }
}

// ---------------------------------------------------------------------------
extern "C" void kernel_launch(void* const* d_in, const int* in_sizes, int n_in,
                              void* d_out, int out_size, void* d_ws, size_t ws_size,
                              hipStream_t stream) {
  (void)in_sizes; (void)n_in; (void)out_size; (void)ws_size;
  const float* q       = (const float*)d_in[0];
  const float* k       = (const float*)d_in[1];
  const float* v       = (const float*)d_in[2];
  const float* proj    = (const float*)d_in[3];
  const int*   sampled = (const int*)d_in[4];
  float* out = (float*)d_out;

  int* q_hash = (int*)d_ws;                 // 1 MB each
  int* k_hash = q_hash + BH * NSEQ;
  int* q_idx  = k_hash + BH * NSEQ;
  int* k_idx  = q_idx  + BH * NSEQ;

  const int rows = BH * NSEQ;
  lsh_hash_kernel<<<rows / 256, 256, 0, stream>>>(q, proj, q_hash);
  lsh_hash_kernel<<<rows / 256, 256, 0, stream>>>(k, proj, k_hash);
  lsh_sort_kernel<<<2 * BH, 256, 0, stream>>>(q_hash, k_hash, q_idx, k_idx);
  hyper_attn_kernel<<<BH * (NSEQ / QTILE), 256, 0, stream>>>(
      q, k, v, sampled, q_idx, k_idx, out);
}
